// BinConv_9491877724618
// MI455X (gfx1250) — 8-bit matrix operand order repaired and checked by reading, compiled, not run on hardware
//
#include <hip/hip_runtime.h>
#include <stdint.h>


typedef int   v16i __attribute__((ext_vector_type(16)));
typedef float v8f  __attribute__((ext_vector_type(8)));

#define CIN   128
#define COUT  128
#define Hh    112
#define Ww    112
#define NB    32
#define PSTR  36
#define HALOW 114

__device__ __forceinline__ int a_koff(int j, int h2) {
    return ((j >> 3) << 6) + (((j >> 2) & 1) << 5) + (((j >> 1) & 1) << 4)
         + ((j & 1) << 2) + (h2 << 3);
}

__global__ void prepack_w_kernel(const float* __restrict__ W, uint32_t* __restrict__ Apack) {
    int idx = blockIdx.x * 256 + threadIdx.x;
    if (idx >= 8 * 9 * 32 * 16) return;
    int j    = idx & 15;
    int lane = (idx >> 4) & 31;
    int gt   = idx >> 9;
    int t    = gt % 9;
    int g    = gt / 9;
    int m    = lane & 15;
    int h2   = lane >> 4;
    int cout = g * 16 + m;
    int kh   = t / 3, kw = t % 3;
    int k0   = a_koff(j, h2);
    uint32_t d = 0;
    #pragma unroll
    for (int b = 0; b < 4; ++b) {
        int cin = k0 + b;
        float w = W[((cout * CIN + cin) * 3 + kh) * 3 + kw];
        uint32_t f8 = (w >= 0.0f) ? 0x38u : 0xB8u;
        d |= f8 << (8 * b);
    }
    Apack[idx] = d;
}

__global__ __launch_bounds__(256) void binarize_nhwc_kernel(const float* __restrict__ x,
                                                            uint32_t* __restrict__ xb) {
    __shared__ uint8_t lds[Ww * (CIN + 4)];
    int nh = blockIdx.x;
    const float* xrow = x + (size_t)(nh / Hh) * CIN * Hh * Ww + (size_t)(nh % Hh) * Ww;
    int tid = threadIdx.x;
    int w   = tid % Ww;
    int c0  = tid / Ww;
    if (c0 < 2) {
        for (int c = c0; c < CIN; c += 2) {
            float v = xrow[(size_t)c * Hh * Ww + w];
            lds[w * (CIN + 4) + c] = (v >= 0.0f) ? 0x38 : 0xB8;
        }
    }
    __syncthreads();
    const uint32_t* l32 = (const uint32_t*)lds;
    uint32_t* orow = xb + (size_t)nh * (Ww * CIN / 4);
    for (int d = tid; d < Ww * CIN / 4; d += 256) {
        int ww = d >> 5, c4 = d & 31;
        orow[d] = l32[ww * 33 + c4];
    }
}

__global__ __launch_bounds__(256) void binconv_wmma_kernel(const uint32_t* __restrict__ xb,
                                                           const uint32_t* __restrict__ Apack,
                                                           const float* __restrict__ bias,
                                                           float* __restrict__ out) {
    __shared__ uint32_t ldsA[3 * HALOW * PSTR];
    int blk = blockIdx.x;
    int h0  = blk % Hh;
    int n   = blk / Hh;
    int tid = threadIdx.x;

    #pragma unroll
    for (int r = 0; r < 3; ++r) {
        int hh = h0 + r - 1;
        bool rowok = (hh >= 0) && (hh < Hh);
        const uint32_t* src = xb + ((size_t)n * Hh + (rowok ? hh : 0)) * (Ww * 32);
        for (int d = tid; d < HALOW * 32; d += 256) {
            int q  = d >> 5;
            int c4 = d & 31;
            int ww = q - 1;
            uint32_t v = 0;
            if (rowok && ww >= 0 && ww < Ww)
                v = src[(size_t)ww * 32 + c4];
            ldsA[(r * HALOW + q) * PSTR + c4] = v;
        }
    }
    __syncthreads();

    int wave = tid >> 5;
    int lane = tid & 31;
    int pcol = lane & 15;
    int h2   = lane >> 4;

    v8f acc[7];
    #pragma unroll
    for (int s = 0; s < 7; ++s) acc[s] = (v8f){};

    const uint32_t* ap = Apack + ((size_t)wave * 9) * 512 + (size_t)lane * 16;

    #pragma unroll 1
    for (int t = 0; t < 9; ++t) {
        int kh = t / 3, kw = t % 3;
        v16i a;
        #pragma unroll
        for (int j = 0; j < 16; ++j)
            a[j] = (int)ap[t * 512 + j];
        int base = (kh * HALOW + kw + pcol) * PSTR;
        #pragma unroll
        for (int s = 0; s < 7; ++s) {
            v16i b;
            #pragma unroll
            for (int j = 0; j < 16; ++j) {
                int k4 = ((j >> 2) << 3) + (((j >> 1) & 1) << 2) + (h2 << 1) + (j & 1);
                b[j] = (int)ldsA[base + s * (16 * PSTR) + k4];
            }
            acc[s] = __builtin_amdgcn_wmma_f32_16x16x128_fp8_fp8(a, b, (short)0, acc[s],
                                                                 false, false);
        }
    }

    float bv[8];
    #pragma unroll
    for (int r = 0; r < 8; ++r) bv[r] = bias[wave * 16 + r + h2 * 8];

    size_t obase = (((size_t)n * COUT + wave * 16 + h2 * 8) * Hh + h0) * Ww + pcol;
    #pragma unroll
    for (int s = 0; s < 7; ++s) {
        #pragma unroll
        for (int r = 0; r < 8; ++r)
            out[obase + (size_t)r * Hh * Ww + s * 16] = acc[s][r] + bv[r];
    }
}

extern "C" void kernel_launch(void* const* d_in, const int* in_sizes, int n_in,
                              void* d_out, int out_size, void* d_ws, size_t ws_size,
                              hipStream_t stream) {
    const float* x = (const float*)d_in[0];
    const float* W = (const float*)d_in[1];
    const float* b = (const float*)d_in[2];
    float* out = (float*)d_out;

    uint32_t* Apack = (uint32_t*)d_ws;
    uint32_t* xb    = (uint32_t*)((char*)d_ws + (256 * 1024));

    prepack_w_kernel<<<(8 * 9 * 32 * 16 + 255) / 256, 256, 0, stream>>>(W, Apack);
    binarize_nhwc_kernel<<<NB * Hh, 256, 0, stream>>>(x, xb);
    binconv_wmma_kernel<<<NB * Hh, 256, 0, stream>>>(xb, Apack, b, out);
}
